// TransformerEncoderLayer_24489903522409
// MI455X (gfx1250) — compile-verified, matrix operand order checked by reading, not run on hardware
//
#include <hip/hip_runtime.h>

typedef __attribute__((ext_vector_type(16))) _Float16 v16h;
typedef __attribute__((ext_vector_type(8)))  _Float16 v8h;
typedef __attribute__((ext_vector_type(8)))  float    v8f;

constexpr int Bn = 4, Sn = 2048, Dn = 512, Hn = 8, DKn = 64, Fn = 2048, Kn = 9, PADn = 4;
constexpr int WAVES = 8;           // waves per block (wave32 -> 256 threads)
constexpr int TPB   = WAVES * 32;

// ---------------------------------------------------------------------------
// WMMA helpers (wave32, 16x16x32 f16 -> f32)
// ---------------------------------------------------------------------------
__device__ __forceinline__ v8f wmma_f16(v16h a, v16h b, v8f c) {
  return __builtin_amdgcn_wmma_f32_16x16x32_f16(false, a, false, b, (short)0, c, false, false);
}

// Fragment loaders for the CDNA5 16-bit A/B VGPR layout.
// lane l: idx = l&15 (row for A / col for B), hi = l>>4;
// halves [0,8) = k in [hi*8, hi*8+8); halves [8,16) = k in [16+hi*8, 16+hi*8+8).
// Both 8-half segments are contiguous -> two explicit 16-byte vector loads.
__device__ __forceinline__ v16h frag_row_vec(const _Float16* __restrict__ p, int hi) {
  const _Float16* q = p + hi * 8;
  const v8h lo = *(const v8h*)(q);
  const v8h hv = *(const v8h*)(q + 16);
  return __builtin_shufflevector(lo, hv, 0, 1, 2, 3, 4, 5, 6, 7,
                                          8, 9, 10, 11, 12, 13, 14, 15);
}
__device__ __forceinline__ v16h frag_ld_vec(const _Float16* __restrict__ base,
                                            long lane_stride, int lane) {
  return frag_row_vec(base + (long)(lane & 15) * lane_stride, lane >> 4);
}

// ---------------------------------------------------------------------------
// fp32 -> fp16 cast (plain)
// ---------------------------------------------------------------------------
__global__ void cast_f32_to_f16(const float* __restrict__ in, _Float16* __restrict__ out, long n) {
  long i = (long)blockIdx.x * blockDim.x + threadIdx.x;
  const long stride = (long)gridDim.x * blockDim.x;
  for (; i < n; i += stride) out[i] = (_Float16)in[i];
}

// Conv weight cast + transpose: (Cout, Cin, K) f32 -> (K, Cout, Cin) f16.
__global__ void cast_transpose_w(const float* __restrict__ in, _Float16* __restrict__ out,
                                 int Cout, int Cin) {
  long i = (long)blockIdx.x * blockDim.x + threadIdx.x;
  const long n = (long)Cout * Cin * Kn;
  const long stride = (long)gridDim.x * blockDim.x;
  for (; i < n; i += stride) {
    const int ci = (int)(i % Cin);
    const long t = i / Cin;
    const int co = (int)(t % Cout);
    const int k  = (int)(t / Cout);
    out[i] = (_Float16)in[((long)co * Cin + ci) * Kn + k];
  }
}

__global__ void zero_f16(_Float16* __restrict__ p, int n) {
  const int i = blockIdx.x * blockDim.x + threadIdx.x;
  if (i < n) p[i] = (_Float16)0.f;
}

// ---------------------------------------------------------------------------
// GEMM: C[M,N] = A_f16[M,K] * W_f16[N,K]^T + bias[N]
// One wave computes a 32x64 tile (2 row-tiles share the 4 B fragments).
// MODE 0: f32 row-major. MODE 1: f16 (B,H,S,DK). MODE 2: f16 V^T (B,H,DK,S).
// ---------------------------------------------------------------------------
template <int MODE>
__global__ void gemm_wmma_kernel(const _Float16* __restrict__ A, const _Float16* __restrict__ W,
                                 const float* __restrict__ bias, float* __restrict__ outF,
                                 _Float16* __restrict__ outH, int M, int N, int Kd) {
  const int lane = threadIdx.x & 31;
  const int wave = threadIdx.x >> 5;
  const int gw = blockIdx.x * WAVES + wave;
  const int ng = N / 64;
  const int mt = gw / ng, nt = gw % ng;
  const int row0 = mt * 32, col0 = nt * 64;
  if (row0 >= M) return;

  v8f acc[2][4] = {};
  for (int k0 = 0; k0 < Kd; k0 += 32) {
    const v16h a0 = frag_ld_vec(A + (size_t)row0 * Kd + k0, Kd, lane);
    const v16h a1 = frag_ld_vec(A + (size_t)(row0 + 16) * Kd + k0, Kd, lane);
#pragma unroll
    for (int t = 0; t < 4; ++t) {
      const v16h bm = frag_ld_vec(W + (size_t)(col0 + t * 16) * Kd + k0, Kd, lane);
      acc[0][t] = wmma_f16(a0, bm, acc[0][t]);
      acc[1][t] = wmma_f16(a1, bm, acc[1][t]);
    }
  }

  const int nl = lane & 15, hi = lane >> 4;
#pragma unroll
  for (int p = 0; p < 2; ++p)
#pragma unroll
    for (int t = 0; t < 4; ++t)
#pragma unroll
      for (int r = 0; r < 8; ++r) {
        const int row = row0 + p * 16 + r + 8 * hi;
        const int col = col0 + t * 16 + nl;
        const float v = acc[p][t][r] + bias[col];
        if (MODE == 0) {
          outF[(size_t)row * N + col] = v;
        } else if (MODE == 1) {
          const int b = row / Sn, s = row % Sn;
          const int h = col >> 6, dk = col & 63;
          outH[(((size_t)(b * Hn + h)) * Sn + s) * DKn + dk] = (_Float16)v;
        } else {
          const int b = row / Sn, s = row % Sn;
          const int h = col >> 6, dk = col & 63;
          outH[(((size_t)(b * Hn + h)) * DKn + dk) * Sn + s] = (_Float16)v;
        }
      }
}

// ---------------------------------------------------------------------------
// Flash-attention: one wave = 16 query rows of one (b,h); 32-key chunks.
// Q tile staged in LDS; V consumed transposed (B,H,DK,S) so every fragment
// in the kernel is two contiguous b128 loads.
// ---------------------------------------------------------------------------
__global__ void attn_wmma_kernel(const _Float16* __restrict__ Q, const _Float16* __restrict__ K,
                                 const _Float16* __restrict__ VT,
                                 const unsigned char* __restrict__ mask,
                                 _Float16* __restrict__ ctx) {
  __shared__ _Float16 ldsQ[WAVES][16 * 64];
  __shared__ _Float16 ldsP[WAVES][16 * 32];
  const int lane = threadIdx.x & 31;
  const int wave = threadIdx.x >> 5;
  const int gw = blockIdx.x * WAVES + wave;
  const int qtiles = Sn / 16;
  const int qt = gw % qtiles;
  const int bh = gw / qtiles;
  const int b = bh / Hn;
  const int h = bh % Hn;
  const _Float16* Qp = Q + (size_t)bh * Sn * DKn + (size_t)qt * 16 * DKn;
  const _Float16* Kp = K + (size_t)bh * Sn * DKn;
  const _Float16* Vp = VT + (size_t)bh * Sn * DKn;   // (DK, S) per head
  const unsigned char* mp = mask + (size_t)b * Sn;
  const int nl = lane & 15, hi = lane >> 4;

  // stage this wave's 16x64 Q tile: each lane copies 32 contiguous halves
  {
    const int r = lane >> 1, c0 = (lane & 1) * 32;
    const _Float16* s = Qp + (size_t)r * DKn + c0;
    _Float16* d = &ldsQ[wave][r * 64 + c0];
#pragma unroll
    for (int j = 0; j < 32; ++j) d[j] = s[j];
  }
  __syncthreads();

  v8f o0 = {}, o1 = {}, o2 = {}, o3 = {};   // 16x64 f32 output accumulator
  float mrow[8], lrow[8];
#pragma unroll
  for (int r = 0; r < 8; ++r) { mrow[r] = -3.0e38f; lrow[r] = 0.f; }

  for (int kc = 0; kc < Sn; kc += 32) {
    // prefetch next chunk of K (streams once through L2)
    __builtin_prefetch(Kp + (size_t)(kc + 32) * DKn, 0, 3);

    const v16h aq0 = frag_ld_vec(&ldsQ[wave][0], 64, lane);
    const v16h aq1 = frag_ld_vec(&ldsQ[wave][32], 64, lane);

    // scores: S0 = keys [kc, kc+16), S1 = keys [kc+16, kc+32); B cols = K rows
    v8f s0 = {}, s1 = {};
    v16h bk;
    bk = frag_ld_vec(Kp + (size_t)kc * DKn,             DKn, lane); s0 = wmma_f16(aq0, bk, s0);
    bk = frag_ld_vec(Kp + (size_t)kc * DKn + 32,        DKn, lane); s0 = wmma_f16(aq1, bk, s0);
    bk = frag_ld_vec(Kp + (size_t)(kc + 16) * DKn,      DKn, lane); s1 = wmma_f16(aq0, bk, s1);
    bk = frag_ld_vec(Kp + (size_t)(kc + 16) * DKn + 32, DKn, lane); s1 = wmma_f16(aq1, bk, s1);

    const float mb0 = mp[kc + nl]      ? 0.f : -1.0e9f;
    const float mb1 = mp[kc + 16 + nl] ? 0.f : -1.0e9f;

#pragma unroll
    for (int r = 0; r < 8; ++r) {
      float e0 = s0[r] * 0.125f + mb0;    // 1/sqrt(64)
      float e1 = s1[r] * 0.125f + mb1;
      float mx = fmaxf(e0, e1);
#pragma unroll
      for (int off = 8; off >= 1; off >>= 1) mx = fmaxf(mx, __shfl_xor(mx, off, 16));
      const float mnew = fmaxf(mrow[r], mx);
      const float corr = __expf(mrow[r] - mnew);
      mrow[r] = mnew;
      const float p0 = __expf(e0 - mnew);
      const float p1 = __expf(e1 - mnew);
      float ps = p0 + p1;
#pragma unroll
      for (int off = 8; off >= 1; off >>= 1) ps += __shfl_xor(ps, off, 16);
      lrow[r] = lrow[r] * corr + ps;
      o0[r] *= corr; o1[r] *= corr; o2[r] *= corr; o3[r] *= corr;
      // stage P row-major [16][32] so it can be re-read in A-fragment layout
      ldsP[wave][(r + 8 * hi) * 32 + nl]      = (_Float16)p0;
      ldsP[wave][(r + 8 * hi) * 32 + 16 + nl] = (_Float16)p1;
    }
    __syncthreads();
    const v16h ap = frag_ld_vec(&ldsP[wave][0], 32, lane);
    v16h bv;  // B[key, dk] from V^T: lane = dk (row of VT, stride Sn), k = key contiguous
    bv = frag_ld_vec(Vp + (size_t)0  * Sn + kc, Sn, lane); o0 = wmma_f16(ap, bv, o0);
    bv = frag_ld_vec(Vp + (size_t)16 * Sn + kc, Sn, lane); o1 = wmma_f16(ap, bv, o1);
    bv = frag_ld_vec(Vp + (size_t)32 * Sn + kc, Sn, lane); o2 = wmma_f16(ap, bv, o2);
    bv = frag_ld_vec(Vp + (size_t)48 * Sn + kc, Sn, lane); o3 = wmma_f16(ap, bv, o3);
    __syncthreads();
  }

#pragma unroll
  for (int r = 0; r < 8; ++r) {
    const int srow = qt * 16 + r + 8 * hi;
    const float inv = 1.0f / lrow[r];
    const size_t base = ((size_t)(b * Sn + srow)) * Dn + (size_t)h * DKn;
    ctx[base + 0  + nl] = (_Float16)(o0[r] * inv);
    ctx[base + 16 + nl] = (_Float16)(o1[r] * inv);
    ctx[base + 32 + nl] = (_Float16)(o2[r] * inv);
    ctx[base + 48 + nl] = (_Float16)(o3[r] * inv);
  }
}

// ---------------------------------------------------------------------------
// Residual add + LayerNorm over D=512. One block (256 thr) per row.
// ---------------------------------------------------------------------------
__global__ void add_ln_kernel(const float* __restrict__ resid, const float* __restrict__ y,
                              const float* __restrict__ gamma, const float* __restrict__ beta,
                              float* __restrict__ outF, _Float16* __restrict__ outH) {
  __shared__ float red[256];
  const int row = blockIdx.x;
  const int tid = threadIdx.x;
  const float* a = resid + (size_t)row * Dn;
  const float* b = y + (size_t)row * Dn;
  const float v0 = a[tid] + b[tid];
  const float v1 = a[tid + 256] + b[tid + 256];
  red[tid] = v0 + v1;
  __syncthreads();
  for (int o = 128; o > 0; o >>= 1) { if (tid < o) red[tid] += red[tid + o]; __syncthreads(); }
  const float mean = red[0] * (1.0f / Dn);
  __syncthreads();
  const float d0 = v0 - mean, d1 = v1 - mean;
  red[tid] = d0 * d0 + d1 * d1;
  __syncthreads();
  for (int o = 128; o > 0; o >>= 1) { if (tid < o) red[tid] += red[tid + o]; __syncthreads(); }
  const float rstd = rsqrtf(red[0] * (1.0f / Dn) + 1e-5f);
  const float r0 = d0 * rstd * gamma[tid] + beta[tid];
  const float r1 = d1 * rstd * gamma[tid + 256] + beta[tid + 256];
  outF[(size_t)row * Dn + tid] = r0;
  outF[(size_t)row * Dn + tid + 256] = r1;
  if (outH) {
    outH[(size_t)row * Dn + tid] = (_Float16)r0;
    outH[(size_t)row * Dn + tid + 256] = (_Float16)r1;
  }
}

// ---------------------------------------------------------------------------
// K=9 conv1d as implicit GEMM. X: (B,S,Cin) f16. Wt: (K, Cout, Cin) f16.
// zrow: Cin zeros for padding rows. One wave: 32 rows x 64 cols.
// ---------------------------------------------------------------------------
template <bool RELU_TO_HALF>
__global__ void conv_wmma_kernel(const _Float16* __restrict__ X, const _Float16* __restrict__ Wt,
                                 const float* __restrict__ bias, const _Float16* __restrict__ zrow,
                                 _Float16* __restrict__ outH, float* __restrict__ outF,
                                 int Cin, int Cout) {
  const int lane = threadIdx.x & 31;
  const int wave = threadIdx.x >> 5;
  const int gw = blockIdx.x * WAVES + wave;
  const int ng = Cout / 64;
  const int mt = gw / ng, nt = gw % ng;
  const int row0 = mt * 32, col0 = nt * 64;
  const int nl = lane & 15, hi = lane >> 4;
  const int ar0 = row0 + nl;            // lane's row, first tile
  const int ar1 = row0 + 16 + nl;       // lane's row, second tile
  const int b0 = ar0 / Sn, s0i = ar0 % Sn;
  const int b1 = ar1 / Sn, s1i = ar1 % Sn;

  v8f acc[2][4] = {};
  for (int k = 0; k < Kn; ++k) {
    const int sk0 = s0i + k - PADn;
    const int sk1 = s1i + k - PADn;
    const _Float16* rowp0 = (sk0 >= 0 && sk0 < Sn) ? (X + ((long)b0 * Sn + sk0) * (long)Cin) : zrow;
    const _Float16* rowp1 = (sk1 >= 0 && sk1 < Sn) ? (X + ((long)b1 * Sn + sk1) * (long)Cin) : zrow;
    const _Float16* wk = Wt + ((size_t)k * Cout + col0) * Cin;
    for (int d0 = 0; d0 < Cin; d0 += 32) {
      const v16h a0 = frag_row_vec(rowp0 + d0, hi);
      const v16h a1 = frag_row_vec(rowp1 + d0, hi);
#pragma unroll
      for (int t = 0; t < 4; ++t) {
        const v16h bm = frag_ld_vec(wk + (size_t)(t * 16) * Cin + d0, Cin, lane);
        acc[0][t] = wmma_f16(a0, bm, acc[0][t]);
        acc[1][t] = wmma_f16(a1, bm, acc[1][t]);
      }
    }
  }

#pragma unroll
  for (int p = 0; p < 2; ++p)
#pragma unroll
    for (int t = 0; t < 4; ++t)
#pragma unroll
      for (int r = 0; r < 8; ++r) {
        const int row = row0 + p * 16 + r + 8 * hi;
        const int col = col0 + t * 16 + nl;
        const float v = acc[p][t][r] + bias[col];
        if (RELU_TO_HALF) outH[(size_t)row * Cout + col] = (_Float16)fmaxf(v, 0.f);
        else              outF[(size_t)row * Cout + col] = v;
      }
}

// ---------------------------------------------------------------------------
// Launch
// ---------------------------------------------------------------------------
extern "C" void kernel_launch(void* const* d_in, const int* in_sizes, int n_in,
                              void* d_out, int out_size, void* d_ws, size_t ws_size,
                              hipStream_t stream) {
  (void)in_sizes; (void)n_in; (void)out_size; (void)ws_size;
  const float* src = (const float*)d_in[0];
  const unsigned char* mask = (const unsigned char*)d_in[1];
  const float* wq = (const float*)d_in[2];  const float* bq = (const float*)d_in[3];
  const float* wk = (const float*)d_in[4];  const float* bk = (const float*)d_in[5];
  const float* wv = (const float*)d_in[6];  const float* bv = (const float*)d_in[7];
  const float* wo = (const float*)d_in[8];  const float* bo = (const float*)d_in[9];
  const float* w1 = (const float*)d_in[10]; const float* b1c = (const float*)d_in[11];
  const float* w2 = (const float*)d_in[12]; const float* b2c = (const float*)d_in[13];
  const float* g1 = (const float*)d_in[14]; const float* be1 = (const float*)d_in[15];
  const float* g2 = (const float*)d_in[16]; const float* be2 = (const float*)d_in[17];

  char* p = (char*)d_ws;
  auto alloc = [&](size_t bytes) { char* r = p; p += (bytes + 255) & ~(size_t)255; return r; };
  const size_t NTOK = (size_t)Bn * Sn;  // 8192 rows

  _Float16* src_h = (_Float16*)alloc(NTOK * Dn * 2);
  _Float16* wq_h  = (_Float16*)alloc((size_t)Dn * Dn * 2);
  _Float16* wk_h  = (_Float16*)alloc((size_t)Dn * Dn * 2);
  _Float16* wv_h  = (_Float16*)alloc((size_t)Dn * Dn * 2);
  _Float16* wo_h  = (_Float16*)alloc((size_t)Dn * Dn * 2);
  _Float16* q_h   = (_Float16*)alloc(NTOK * Dn * 2);
  _Float16* k_h   = (_Float16*)alloc(NTOK * Dn * 2);
  _Float16* v_h   = (_Float16*)alloc(NTOK * Dn * 2);     // V^T (B,H,DK,S)
  _Float16* ctx_h = (_Float16*)alloc(NTOK * Dn * 2);
  float*    attn_out = (float*)alloc(NTOK * Dn * 4);
  float*    x_f   = (float*)alloc(NTOK * Dn * 4);
  _Float16* x_h   = (_Float16*)alloc(NTOK * Dn * 2);
  _Float16* w1_h  = (_Float16*)alloc((size_t)Fn * Dn * Kn * 2);   // (K,F,D) layout
  _Float16* w2_h  = (_Float16*)alloc((size_t)Dn * Fn * Kn * 2);   // (K,D,F) layout
  _Float16* h_h   = (_Float16*)alloc(NTOK * Fn * 2);
  _Float16* zrow  = (_Float16*)alloc((size_t)Fn * 2);             // zero padding row
  float*    ffn   = (float*)q_h;   // q/k/v (24MB) dead after attention; reuse for ffn (16MB)

  // staging: casts, transposes, zero row
  cast_f32_to_f16<<<1024, 256, 0, stream>>>(src, src_h, (long)NTOK * Dn);
  cast_f32_to_f16<<<256, 256, 0, stream>>>(wq, wq_h, (long)Dn * Dn);
  cast_f32_to_f16<<<256, 256, 0, stream>>>(wk, wk_h, (long)Dn * Dn);
  cast_f32_to_f16<<<256, 256, 0, stream>>>(wv, wv_h, (long)Dn * Dn);
  cast_f32_to_f16<<<256, 256, 0, stream>>>(wo, wo_h, (long)Dn * Dn);
  cast_transpose_w<<<2048, 256, 0, stream>>>(w1, w1_h, Fn, Dn);
  cast_transpose_w<<<2048, 256, 0, stream>>>(w2, w2_h, Dn, Fn);
  zero_f16<<<Fn / 256, 256, 0, stream>>>(zrow, Fn);

  const dim3 blk(TPB);
  const int gemm_blocks = (int)((NTOK / 32) * (Dn / 64) / WAVES);   // 256

  // Q,K projections -> (B,H,S,DK); V projection -> V^T (B,H,DK,S)
  gemm_wmma_kernel<1><<<gemm_blocks, blk, 0, stream>>>(src_h, wq_h, bq, nullptr, q_h, (int)NTOK, Dn, Dn);
  gemm_wmma_kernel<1><<<gemm_blocks, blk, 0, stream>>>(src_h, wk_h, bk, nullptr, k_h, (int)NTOK, Dn, Dn);
  gemm_wmma_kernel<2><<<gemm_blocks, blk, 0, stream>>>(src_h, wv_h, bv, nullptr, v_h, (int)NTOK, Dn, Dn);

  // attention -> ctx (B,S,D) f16
  const int attn_blocks = (Bn * Hn * (Sn / 16)) / WAVES;            // 512
  attn_wmma_kernel<<<attn_blocks, blk, 0, stream>>>(q_h, k_h, v_h, mask, ctx_h);

  // output projection -> attn_out f32
  gemm_wmma_kernel<0><<<gemm_blocks, blk, 0, stream>>>(ctx_h, wo_h, bo, attn_out, nullptr, (int)NTOK, Dn, Dn);

  // x = LN(src + attn_out)
  add_ln_kernel<<<(int)NTOK, 256, 0, stream>>>(src, attn_out, g1, be1, x_f, x_h);

  // FFN conv1 (relu, f16) and conv2 (f32)
  const int c1_blocks = (int)((NTOK / 32) * (Fn / 64) / WAVES);     // 1024
  conv_wmma_kernel<true><<<c1_blocks, blk, 0, stream>>>(x_h, w1_h, b1c, zrow, h_h, nullptr, Dn, Fn);
  const int c2_blocks = (int)((NTOK / 32) * (Dn / 64) / WAVES);     // 256
  conv_wmma_kernel<false><<<c2_blocks, blk, 0, stream>>>(h_h, w2_h, b2c, zrow, nullptr, ffn, Fn, Dn);

  // out = LN(x + ffn)
  add_ln_kernel<<<(int)NTOK, 256, 0, stream>>>(x_f, ffn, g2, be2, (float*)d_out, nullptr);
}
